// GraphGatedEncoder_32341103738941
// MI455X (gfx1250) — matrix operand order repaired and checked by reading, not compiled, not run on hardware
//
#include <hip/hip_runtime.h>

#define HD   256                 // hidden size
#define HD3  768                 // 3*H
#define MROWS 32                 // nodes per GRU block (2 WMMA row tiles)
#define GATE_STRIDE (256 * HD)   // elems per gate block in Wcat

typedef __attribute__((ext_vector_type(16))) __bf16 bf16x16;
typedef __attribute__((ext_vector_type(8)))  float  floatx8;

// ---------------------------------------------------------------- helpers
__device__ __forceinline__ bf16x16 ld_frag(const __bf16* p0, const __bf16* p1) {
  union { uint4 q[2]; bf16x16 v; } t;
  t.q[0] = *(const uint4*)p0;
  t.q[1] = *(const uint4*)p1;
  return t.v;
}

__device__ __forceinline__ floatx8 wmma_bf16(bf16x16 a, bf16x16 b, floatx8 c) {
  // v_wmma_f32_16x16x32_bf16 : D = A(16x32) * B(32x16) + C(16x16 f32)
  return __builtin_amdgcn_wmma_f32_16x16x32_bf16(
      /*neg_a=*/false, a, /*neg_b=*/false, b,
      /*c_mod=*/(short)0, c, /*reuse_a=*/false, /*reuse_b=*/false);
}

__device__ __forceinline__ float sigmoid_f(float x) {
  return 1.0f / (1.0f + __expf(-x));
}
__device__ __forceinline__ float tanh_f(float x) {
  x = fminf(fmaxf(x, -15.0f), 15.0f);
  float e = __expf(2.0f * x);
  return (e - 1.0f) / (e + 1.0f);
}

// ---------------------------------------------------------------- prep
// out[n*512 + c] = x ;  out[n*512 + 256 + c] = x (h init) ; h_bf = bf16(x)
__global__ void __launch_bounds__(256)
k_prep(const float* __restrict__ x, float* __restrict__ out,
       __bf16* __restrict__ h_bf, long long nh) {
  long long i = (long long)blockIdx.x * blockDim.x + threadIdx.x;
  if (i >= nh) return;
  long long node = i >> 8;
  int c = (int)(i & 255);
  float v = x[i];
  out[node * 512 + c]       = v;
  out[node * 512 + 256 + c] = v;
  h_bf[i] = (__bf16)v;
}

__global__ void __launch_bounds__(256)
k_invdeg(const float* __restrict__ nn, float* __restrict__ invd, int n) {
  int i = blockIdx.x * blockDim.x + threadIdx.x;
  if (i < n) invd[i] = 1.0f / (nn[i] + 1e-7f);
}

// Wcat rows 0..767 = W_ih (r,z,n), rows 768..1535 = W_hh (r,z,n); row-major K=256
__global__ void __launch_bounds__(256)
k_wconv(const float* __restrict__ wih, const float* __restrict__ whh,
        __bf16* __restrict__ wcat) {
  int i = blockIdx.x * blockDim.x + threadIdx.x;   // 0 .. 2*768*256-1
  float v = (i < HD3 * HD) ? wih[i] : whh[i - HD3 * HD];
  wcat[i] = (__bf16)v;
}

__global__ void __launch_bounds__(256)
k_zero4(float4* __restrict__ p, long long n4) {
  long long i = (long long)blockIdx.x * blockDim.x + threadIdx.x;
  if (i < n4) p[i] = make_float4(0.f, 0.f, 0.f, 0.f);
}

// ---------------------------------------------------------------- SpMM scatter
// one wave per edge; lane handles 8 channels; h lives in d_out rows (stride 512, +256)
__global__ void __launch_bounds__(256)
k_spmm(const float* __restrict__ hrows,   // = d_out base
       const int* __restrict__ esrc, const int* __restrict__ edst,
       const float* __restrict__ eval, float* __restrict__ u, int E) {
  const int wid  = (int)(((long long)blockIdx.x * 256 + threadIdx.x) >> 5);
  const int lane = threadIdx.x & 31;
  if (wid >= E) return;
  const int   s = esrc[wid];
  const int   d = edst[wid];
  const float v = eval[wid];
  const float4* hp = (const float4*)(hrows + (size_t)s * 512 + 256) + (lane << 1);
  float4 a = hp[0];
  float4 b = hp[1];
  float* up = u + (size_t)d * HD + (lane << 3);
  atomicAdd(up + 0, a.x * v); atomicAdd(up + 1, a.y * v);
  atomicAdd(up + 2, a.z * v); atomicAdd(up + 3, a.w * v);
  atomicAdd(up + 4, b.x * v); atomicAdd(up + 5, b.y * v);
  atomicAdd(up + 6, b.z * v); atomicAdd(up + 7, b.w * v);
}

// u *= inv_deg ; u_bf = bf16(u)
__global__ void __launch_bounds__(256)
k_scale(const float* __restrict__ u, const float* __restrict__ invd,
        __bf16* __restrict__ ub, long long nh) {
  long long i = (long long)blockIdx.x * blockDim.x + threadIdx.x;
  if (i < nh) ub[i] = (__bf16)(u[i] * invd[i >> 8]);
}

// ---------------------------------------------------------------- fused GRU (WMMA)
// block = 32 nodes (2 row tiles), 8 waves. Per wave: 2 column-groups of 16
// hidden cols; per column-group 12 f32 accumulators (6 gates x 2 row tiles).
// Every B fragment fetched from L2 feeds TWO WMMAs (M-blocking for weight reuse).
__global__ void __launch_bounds__(256)
k_gru(const __bf16* __restrict__ u_bf, __bf16* h_bf,
      float* out /* d_out: h at [row*512+256+col] */,
      const __bf16* __restrict__ Wcat,
      const float* __restrict__ b_ih, const float* __restrict__ b_hh) {
  __shared__ __bf16 uA[MROWS * HD];
  __shared__ __bf16 hA[MROWS * HD];
  const int tile = blockIdx.x;          // 32-row node tile
  const int tid  = threadIdx.x;

  { // stage A tiles (32x256 bf16 each = 16 KB) into LDS
    const size_t gbase = (size_t)tile * (MROWS * HD);
    const uint4* gu = (const uint4*)(u_bf + gbase);
    const uint4* gh = (const uint4*)(h_bf + gbase);
    uint4* su = (uint4*)uA;
    uint4* sh = (uint4*)hA;
#pragma unroll
    for (int j = 0; j < 4; ++j) {       // 1024 uint4 per matrix / 256 threads
      su[tid + j * 256] = gu[tid + j * 256];
      sh[tid + j * 256] = gh[tid + j * 256];
    }
  }
  __syncthreads();

  const int wave  = tid >> 5;
  const int lane  = tid & 31;
  const int lhalf = lane >> 4;    // 0/1: which K/M half this lane covers
  const int ln    = lane & 15;

  for (int half = 0; half < 2; ++half) {
    const int cg  = wave + half * 8;     // column group 0..15
    const int col = cg * 16 + ln;        // this lane's hidden column (N index)

    // B operand: lane ln = column, chunks at k0+lhalf*8 and k0+16+lhalf*8
    const __bf16* wb   = Wcat + (size_t)col * HD + lhalf * 8;
    const __bf16* w_ir = wb + 0 * GATE_STRIDE;
    const __bf16* w_iz = wb + 1 * GATE_STRIDE;
    const __bf16* w_in = wb + 2 * GATE_STRIDE;
    const __bf16* w_hr = wb + 3 * GATE_STRIDE;
    const __bf16* w_hz = wb + 4 * GATE_STRIDE;
    const __bf16* w_hn = wb + 5 * GATE_STRIDE;

    // A operand: lane ln = row M, chunks at k0+lhalf*8 and k0+16+lhalf*8
    const __bf16* au0 = &uA[(ln)      * HD + lhalf * 8];   // row tile 0
    const __bf16* au1 = &uA[(ln + 16) * HD + lhalf * 8];   // row tile 1
    const __bf16* ah0 = &hA[(ln)      * HD + lhalf * 8];
    const __bf16* ah1 = &hA[(ln + 16) * HD + lhalf * 8];

    floatx8 ir0 = {}, iz0 = {}, in0 = {}, hr0 = {}, hz0 = {}, hn0 = {};
    floatx8 ir1 = {}, iz1 = {}, in1 = {}, hr1 = {}, hz1 = {}, hn1 = {};

#pragma unroll
    for (int k0 = 0; k0 < HD; k0 += 32) {
      bf16x16 bir = ld_frag(w_ir + k0, w_ir + k0 + 16);
      bf16x16 biz = ld_frag(w_iz + k0, w_iz + k0 + 16);
      bf16x16 bin = ld_frag(w_in + k0, w_in + k0 + 16);
      bf16x16 bhr = ld_frag(w_hr + k0, w_hr + k0 + 16);
      bf16x16 bhz = ld_frag(w_hz + k0, w_hz + k0 + 16);
      bf16x16 bhn = ld_frag(w_hn + k0, w_hn + k0 + 16);
      bf16x16 a0  = ld_frag(au0 + k0, au0 + k0 + 16);
      bf16x16 a1  = ld_frag(au1 + k0, au1 + k0 + 16);
      bf16x16 c0  = ld_frag(ah0 + k0, ah0 + k0 + 16);
      bf16x16 c1  = ld_frag(ah1 + k0, ah1 + k0 + 16);
      ir0 = wmma_bf16(a0, bir, ir0);  ir1 = wmma_bf16(a1, bir, ir1);
      iz0 = wmma_bf16(a0, biz, iz0);  iz1 = wmma_bf16(a1, biz, iz1);
      in0 = wmma_bf16(a0, bin, in0);  in1 = wmma_bf16(a1, bin, in1);
      hr0 = wmma_bf16(c0, bhr, hr0);  hr1 = wmma_bf16(c1, bhr, hr1);
      hz0 = wmma_bf16(c0, bhz, hz0);  hz1 = wmma_bf16(c1, bhz, hz1);
      hn0 = wmma_bf16(c0, bhn, hn0);  hn1 = wmma_bf16(c1, bhn, hn1);
    }

    // gate math; C/D layout: lane -> N (=col), VGPR v -> M = v + lhalf*8
    const float br  = b_ih[col]          + b_hh[col];
    const float bz  = b_ih[HD + col]     + b_hh[HD + col];
    const float bni = b_ih[2 * HD + col];
    const float bnh = b_hh[2 * HD + col];

#pragma unroll
    for (int mt = 0; mt < 2; ++mt) {
      const floatx8 air = mt ? ir1 : ir0;
      const floatx8 aiz = mt ? iz1 : iz0;
      const floatx8 ain = mt ? in1 : in0;
      const floatx8 ahr = mt ? hr1 : hr0;
      const floatx8 ahz = mt ? hz1 : hz0;
      const floatx8 ahn = mt ? hn1 : hn0;
      const int rowbase = tile * MROWS + mt * 16 + lhalf * 8;
#pragma unroll
      for (int v = 0; v < 8; ++v) {
        const size_t row  = (size_t)(rowbase + v);
        const size_t oidx = row * 512 + 256 + col;
        const float hold  = out[oidx];
        const float r = sigmoid_f(air[v] + ahr[v] + br);
        const float z = sigmoid_f(aiz[v] + ahz[v] + bz);
        const float n = tanh_f(ain[v] + bni + r * (ahn[v] + bnh));
        const float hnew = (1.0f - z) * n + z * hold;
        out[oidx] = hnew;
        h_bf[row * HD + col] = (__bf16)hnew;
      }
    }
  }
}

// ---------------------------------------------------------------- launch
extern "C" void kernel_launch(void* const* d_in, const int* in_sizes, int n_in,
                              void* d_out, int out_size, void* d_ws, size_t ws_size,
                              hipStream_t stream) {
  const float* x     = (const float*)d_in[0];
  const int*   esrc  = (const int*)  d_in[1];
  const int*   edst  = (const int*)  d_in[2];
  const float* eval  = (const float*)d_in[3];
  const float* nn    = (const float*)d_in[4];
  const float* W_ih  = (const float*)d_in[5];
  const float* W_hh  = (const float*)d_in[6];
  const float* b_ih  = (const float*)d_in[7];
  const float* b_hh  = (const float*)d_in[8];

  const int N = in_sizes[0] / HD;       // 100000 (multiple of 32)
  const int E = in_sizes[1];            // 3.2M
  const long long NH = (long long)N * HD;

  // workspace carve-up (all offsets 256B aligned)
  char* ws = (char*)d_ws;
  size_t off = 0;
  float*  u_f32 = (float*) (ws + off); off += (size_t)NH * 4;          // 102.4 MB
  __bf16* h_bf  = (__bf16*)(ws + off); off += (size_t)NH * 2;          //  51.2 MB
  __bf16* u_bf  = (__bf16*)(ws + off); off += (size_t)NH * 2;          //  51.2 MB
  __bf16* Wcat  = (__bf16*)(ws + off); off += (size_t)2 * HD3 * HD * 2;//  0.79 MB
  float*  invd  = (float*) (ws + off); off += (size_t)N * 4;           //  0.4 MB
  (void)ws_size;

  float* out = (float*)d_out;           // [N,512]; h lives in [:,256:512]

  const int T = 256;
  const int gNH = (int)((NH + T - 1) / T);

  // per-launch prep — deterministic
  k_prep  <<<gNH, T, 0, stream>>>(x, out, h_bf, NH);
  k_invdeg<<<(N + T - 1) / T, T, 0, stream>>>(nn, invd, N);
  k_wconv <<<(2 * HD3 * HD) / T, T, 0, stream>>>(W_ih, W_hh, Wcat);

  const int gSpmm = (E + 7) / 8;        // 8 waves/block, 1 edge/wave
  const int gGru  = N / MROWS;          // 3125 blocks

  for (int step = 0; step < 2; ++step) {
    k_zero4<<<(int)((NH / 4 + T - 1) / T), T, 0, stream>>>((float4*)u_f32, NH / 4);
    k_spmm <<<gSpmm, T, 0, stream>>>(out, esrc, edst, eval, u_f32, E);
    k_scale<<<gNH, T, 0, stream>>>(u_f32, invd, u_bf, NH);
    k_gru  <<<gGru, T, 0, stream>>>(u_bf, h_bf, out, Wcat, b_ih, b_hh);
  }
}
